// TransformerLM_89017492177540
// MI455X (gfx1250) — matrix operand order repaired and checked by reading, not compiled, not run on hardware
//
#include <hip/hip_runtime.h>
#include <math.h>

// ---------------- problem constants (match reference) ----------------
#define Bsz  4
#define Seq  1024
#define Dm   1024
#define Hn   16
#define Ly   12
#define Cv   8192
#define Ffn  4096
#define HDm  64
#define Mrows (Bsz*Seq)  // 4096
#define PADTOK 2

typedef __attribute__((ext_vector_type(16))) __bf16 v16bf;
typedef __attribute__((ext_vector_type(8)))  __bf16 v8bf;
typedef __attribute__((ext_vector_type(8)))  float  v8f;

// fp32 -> bf16 round-to-nearest-even
__device__ inline __bf16 f2bf(float f) {
  unsigned u = __builtin_bit_cast(unsigned, f);
  unsigned r = u + 0x7FFFu + ((u >> 16) & 1u);
  unsigned short h = (unsigned short)(r >> 16);
  return __builtin_bit_cast(__bf16, h);
}

__device__ inline v8f wmma_bf16(v16bf a, v16bf b, v8f c) {
  return __builtin_amdgcn_wmma_f32_16x16x32_bf16(
      false, a, false, b, (short)0, c, false, false);
}

__device__ inline v16bf cat8(v8bf lo, v8bf hi) {
  return __builtin_shufflevector(lo, hi, 0,1,2,3,4,5,6,7,8,9,10,11,12,13,14,15);
}

// CDNA5 async global->LDS copy (16B per lane), tracked by ASYNCcnt
__device__ inline void async_ld_b128(void* lds, const void* g) {
  unsigned l = (unsigned)(uintptr_t)lds;           // LDS offset = low 32 bits
  unsigned long long a = (unsigned long long)(uintptr_t)g;
  asm volatile("global_load_async_to_lds_b128 %0, %1, off" :: "v"(l), "v"(a) : "memory");
}
__device__ inline void wait_async0() {
  asm volatile("s_wait_asynccnt 0x0" ::: "memory");
}

// ---------------- fp32 -> bf16 streaming conversion ----------------
__global__ __launch_bounds__(256) void cvt_kernel(
    const float* __restrict__ in, __bf16* __restrict__ out, int n8) {
  int i = blockIdx.x * 256 + threadIdx.x;   // 8 elements per thread
  if (i < n8) {
    const float4* p = (const float4*)(in + (size_t)i * 8);
    float4 a = p[0], b = p[1];
    v8bf o;
    o[0]=f2bf(a.x); o[1]=f2bf(a.y); o[2]=f2bf(a.z); o[3]=f2bf(a.w);
    o[4]=f2bf(b.x); o[5]=f2bf(b.y); o[6]=f2bf(b.z); o[7]=f2bf(b.w);
    *(v8bf*)(out + (size_t)i * 8) = o;
  }
}

// ---------------- embedding + positional encoding ----------------
__global__ __launch_bounds__(256) void embed_kernel(
    const int* __restrict__ tokens, const float* __restrict__ emb,
    const float* __restrict__ pe, float* __restrict__ X) {
  const int row = blockIdx.x;
  const int s   = row & (Seq - 1);
  const int tok = tokens[row];
  const size_t ro = (size_t)row * Dm, eo = (size_t)tok * Dm, po = (size_t)s * Dm;
#pragma unroll
  for (int i = 0; i < 4; ++i) {
    int d = threadIdx.x + i * 256;
    X[ro + d] = emb[eo + d] * 32.0f + pe[po + d];
  }
}

// ---------------- LayerNorm: f32 in -> bf16 out ----------------
__global__ __launch_bounds__(256) void ln_kernel(
    const float* __restrict__ X, const float* __restrict__ g,
    const float* __restrict__ bta, __bf16* __restrict__ Out) {
  __shared__ float red1[8], red2[8];
  const int tid = threadIdx.x, lane = tid & 31, wid = tid >> 5;
  const size_t ro = (size_t)blockIdx.x * Dm;
  float vals[4], s = 0.f, s2 = 0.f;
#pragma unroll
  for (int i = 0; i < 4; ++i) {
    float v = X[ro + tid + i * 256];
    vals[i] = v; s += v; s2 += v * v;
  }
#pragma unroll
  for (int o = 16; o > 0; o >>= 1) { s += __shfl_xor(s, o, 32); s2 += __shfl_xor(s2, o, 32); }
  if (lane == 0) { red1[wid] = s; red2[wid] = s2; }
  __syncthreads();
  if (tid == 0) {
    float a = 0.f, b = 0.f;
    for (int i = 0; i < 8; ++i) { a += red1[i]; b += red2[i]; }
    red1[0] = a; red2[0] = b;
  }
  __syncthreads();
  const float mu  = red1[0] * (1.0f / Dm);
  const float var = red2[0] * (1.0f / Dm) - mu * mu;
  const float rstd = rsqrtf(var + 1e-5f);
#pragma unroll
  for (int i = 0; i < 4; ++i) {
    int d = tid + i * 256;
    Out[ro + d] = f2bf((vals[i] - mu) * rstd * g[d] + bta[d]);
  }
}

// ---------------- NT GEMM (bf16 in, WMMA, async double-buffered LDS) -------
// C[M,N] = A[M,K] * W[N,K]^T ; flags: 1 bias, 2 gelu, 4 resadd(f32), 8 bf16 out
#define GF_BIAS 1
#define GF_GELU 2
#define GF_RES  4
#define GF_OBF  8
__global__ __launch_bounds__(256) void gemm_nt(
    const __bf16* __restrict__ Ab, const __bf16* __restrict__ Wb,
    const float* __restrict__ bias, float* __restrict__ Cf,
    __bf16* __restrict__ Cb, int M, int N, int K, int flags) {
  __shared__ __bf16 As[2][128][32];
  __shared__ __bf16 Bs[2][128][32];
  const int tid  = threadIdx.x;
  const int lane = tid & 31, wid = tid >> 5;
  const int n16  = lane & 15, half = lane >> 4;
  const int wm   = wid & 3;           // 4 wave-rows of 32
  const int wn   = wid >> 2;          // 2 wave-cols of 64
  const int m0   = blockIdx.y * 128, n0 = blockIdx.x * 128;

  v8f acc[2][4];
#pragma unroll
  for (int mi = 0; mi < 2; ++mi)
#pragma unroll
    for (int ni = 0; ni < 4; ++ni)
#pragma unroll
      for (int r = 0; r < 8; ++r) acc[mi][ni][r] = 0.0f;

  const int e0 = tid * 2;
  auto stage = [&](int kt, int bsel) {
#pragma unroll
    for (int e = 0; e < 2; ++e) {
      int ee = e0 + e;
      int row = ee >> 2, col = (ee & 3) * 8;
      async_ld_b128(&As[bsel][row][col], Ab + (size_t)(m0 + row) * K + kt * 32 + col);
      async_ld_b128(&Bs[bsel][row][col], Wb + (size_t)(n0 + row) * K + kt * 32 + col);
    }
  };

  const int nk = K >> 5;
  stage(0, 0);
  wait_async0();
  __syncthreads();

  for (int ki = 0; ki < nk; ++ki) {
    const int cur = ki & 1;
    if (ki + 1 < nk) stage(ki + 1, cur ^ 1);

    v16bf af[2], bfr[4];
#pragma unroll
    for (int mi = 0; mi < 2; ++mi) {
      const __bf16* pr = &As[cur][wm * 32 + mi * 16 + n16][0];
      v8bf lo = *(const v8bf*)(pr + half * 16);
      v8bf hi = *(const v8bf*)(pr + half * 16 + 8);
      af[mi] = cat8(lo, hi);
    }
#pragma unroll
    for (int ni = 0; ni < 4; ++ni)
      bfr[ni] = *(const v16bf*)(&Bs[cur][wn * 64 + ni * 16 + n16][half * 16]);

#pragma unroll
    for (int mi = 0; mi < 2; ++mi)
#pragma unroll
      for (int ni = 0; ni < 4; ++ni)
        acc[mi][ni] = wmma_bf16(af[mi], bfr[ni], acc[mi][ni]);

    wait_async0();       // next tile landed (overlapped with WMMAs above)
    __syncthreads();
  }

#pragma unroll
  for (int mi = 0; mi < 2; ++mi)
#pragma unroll
    for (int ni = 0; ni < 4; ++ni)
#pragma unroll
      for (int r = 0; r < 8; ++r) {
        int m = m0 + wm * 32 + mi * 16 + r + half * 8;
        int n = n0 + wn * 64 + ni * 16 + n16;
        float v = acc[mi][ni][r];
        if (flags & GF_BIAS) v += bias[n];
        if (flags & GF_GELU) v = 0.5f * v * (1.0f + erff(v * 0.70710678118f));
        size_t idx = (size_t)m * N + n;
        if (flags & GF_RES)      Cf[idx] += v;
        else if (flags & GF_OBF) Cb[idx] = f2bf(v);
        else                     Cf[idx] = v;
      }
}

// ---------------- flash attention (bf16 in/out, causal, online softmax) ----
// grid: (Seq/128, Bsz*Hn). Block = 8 waves sharing async-staged K/V tiles;
// all waves iterate the block-uniform k range, causal mask trims per wave.
__global__ __launch_bounds__(256) void flash_attn(
    const __bf16* __restrict__ Qb, const __bf16* __restrict__ Kb,
    const __bf16* __restrict__ Vb, __bf16* __restrict__ Yb) {
  __shared__ __bf16 ktile[2][32][64];
  __shared__ __bf16 vtile[2][32][64];
  __shared__ __bf16 pbuf[8][16][32];
  const int tid  = threadIdx.x;
  const int wid  = tid >> 5, lane = tid & 31;
  const int n16  = lane & 15, half = lane >> 4;
  const int bh   = blockIdx.y;
  const int b    = bh >> 4, hh = bh & 15;
  const int q0   = blockIdx.x * 128 + wid * 16;
  const size_t base = (size_t)(b * Seq) * Dm + hh * HDm;

  // Q fragments (A layout), vectorized global bf16 loads
  v16bf aq[2];
  {
    const __bf16* qrow = Qb + base + (size_t)(q0 + n16) * Dm;
#pragma unroll
    for (int c = 0; c < 2; ++c) {
      v8bf lo = *(const v8bf*)(qrow + c * 32 + half * 16);
      v8bf hi = *(const v8bf*)(qrow + c * 32 + half * 16 + 8);
      aq[c] = cat8(lo, hi);
    }
  }

  v8f o[4];
  float rmax[8], rsum[8];
#pragma unroll
  for (int c = 0; c < 4; ++c)
#pragma unroll
    for (int r = 0; r < 8; ++r) o[c][r] = 0.0f;
#pragma unroll
  for (int r = 0; r < 8; ++r) { rmax[r] = -INFINITY; rsum[r] = 0.0f; }

  const int srow = tid >> 3, scol = (tid & 7) * 8;   // staging: 32 rows x 64 cols
  auto stageKV = [&](int j, int bsel) {
    int kp = j * 32 + srow;
    kp = (kp < Seq) ? kp : (Seq - 1);
    const size_t go = base + (size_t)kp * Dm + scol;
    async_ld_b128(&ktile[bsel][srow][scol], Kb + go);
    async_ld_b128(&vtile[bsel][srow][scol], Vb + go);
  };

  const int jmax = blockIdx.x * 4 + 3;   // block-uniform trip count
  stageKV(0, 0);
  wait_async0();
  __syncthreads();

  for (int j = 0; j <= jmax; ++j) {
    const int cur = j & 1;
    if (j + 1 <= jmax) stageKV(j + 1, cur ^ 1);
    const int kpos0 = j << 5;

    // ---- S = Q * K^T ----
    v8f sacc[2];
#pragma unroll
    for (int t = 0; t < 2; ++t)
#pragma unroll
      for (int r = 0; r < 8; ++r) sacc[t][r] = 0.0f;
#pragma unroll
    for (int c = 0; c < 2; ++c) {
      v16bf bk0 = *(const v16bf*)(&ktile[cur][n16][c * 32 + half * 16]);
      v16bf bk1 = *(const v16bf*)(&ktile[cur][16 + n16][c * 32 + half * 16]);
      sacc[0] = wmma_bf16(aq[c], bk0, sacc[0]);
      sacc[1] = wmma_bf16(aq[c], bk1, sacc[1]);
    }

    // ---- causal mask + online softmax (scale 1/sqrt(64) applied here) ----
    float p0[8], p1[8];
#pragma unroll
    for (int r = 0; r < 8; ++r) {
      const int m = r + half * 8;
      float s0 = sacc[0][r] * 0.125f, s1 = sacc[1][r] * 0.125f;
      if (kpos0 + n16      > q0 + m) s0 = -1e8f;
      if (kpos0 + 16 + n16 > q0 + m) s1 = -1e8f;
      float mx = fmaxf(s0, s1);
#pragma unroll
      for (int off = 1; off < 16; off <<= 1) mx = fmaxf(mx, __shfl_xor(mx, off, 16));
      const float nm = fmaxf(rmax[r], mx);
      const float alpha = __expf(rmax[r] - nm);
      s0 = __expf(s0 - nm); s1 = __expf(s1 - nm);
      float rs = s0 + s1;
#pragma unroll
      for (int off = 1; off < 16; off <<= 1) rs += __shfl_xor(rs, off, 16);
      rsum[r] = rsum[r] * alpha + rs;
      rmax[r] = nm;
#pragma unroll
      for (int c = 0; c < 4; ++c) o[c][r] *= alpha;
      p0[r] = s0; p1[r] = s1;
    }

    // ---- P: C-layout -> A-layout via per-wave LDS round trip ----
#pragma unroll
    for (int r = 0; r < 8; ++r) {
      pbuf[wid][r + half * 8][n16]      = f2bf(p0[r]);
      pbuf[wid][r + half * 8][16 + n16] = f2bf(p1[r]);
    }
    asm volatile("s_wait_dscnt 0x0" ::: "memory");
    v16bf pf;
    {
      const __bf16* pr = &pbuf[wid][n16][0];
      v8bf lo = *(const v8bf*)(pr + half * 16);
      v8bf hi = *(const v8bf*)(pr + half * 16 + 8);
      pf = cat8(lo, hi);
    }

    // ---- O += P * V ----
#pragma unroll
    for (int c = 0; c < 4; ++c) {
      v16bf bvf;
#pragma unroll
      for (int jj = 0; jj < 16; ++jj)
        bvf[jj] = vtile[cur][half * 16 + jj][c * 16 + n16];
      o[c] = wmma_bf16(pf, bvf, o[c]);
    }

    wait_async0();
    __syncthreads();
  }

  // ---- finalize: O /= rowsum, write bf16 ----
#pragma unroll
  for (int r = 0; r < 8; ++r) {
    const int m = r + half * 8;
    const float inv = 1.0f / rsum[r];
    __bf16* yrow = Yb + base + (size_t)(q0 + m) * Dm;
#pragma unroll
    for (int c = 0; c < 4; ++c) yrow[c * 16 + n16] = f2bf(o[c][r] * inv);
  }
}

// ---------------- final loss: logsumexp + gather + masked sum ----------------
__global__ void zero_out(float* out) { if (threadIdx.x < Bsz) out[threadIdx.x] = 0.0f; }

__global__ __launch_bounds__(256) void loss_kernel(
    const float* __restrict__ logits, const int* __restrict__ tokens,
    float* __restrict__ out) {
  __shared__ float red[8];
  __shared__ float bc;
  const int tid = threadIdx.x, lane = tid & 31, wid = tid >> 5;
  const int row = blockIdx.x;
  const int b = row >> 10;
  const int tok = tokens[row];
  const float* lr = logits + (size_t)row * Cv;

  float mx = -INFINITY;
  for (int c = tid; c < Cv; c += 256) mx = fmaxf(mx, lr[c]);
#pragma unroll
  for (int o = 16; o > 0; o >>= 1) mx = fmaxf(mx, __shfl_xor(mx, o, 32));
  if (lane == 0) red[wid] = mx;
  __syncthreads();
  if (tid == 0) {
    float m = red[0];
    for (int i = 1; i < 8; ++i) m = fmaxf(m, red[i]);
    bc = m;
  }
  __syncthreads();
  mx = bc;

  float sum = 0.0f;
  for (int c = tid; c < Cv; c += 256) sum += __expf(lr[c] - mx);
#pragma unroll
  for (int o = 16; o > 0; o >>= 1) sum += __shfl_xor(sum, o, 32);
  __syncthreads();
  if (lane == 0) red[wid] = sum;
  __syncthreads();
  if (tid == 0) {
    float sacc = 0.0f;
    for (int i = 0; i < 8; ++i) sacc += red[i];
    const float lse = mx + __logf(sacc);
    if (tok != PADTOK) atomicAdd(&out[b], lr[tok] - lse);
  }
}

// ---------------- driver ----------------
extern "C" void kernel_launch(void* const* d_in, const int* in_sizes, int n_in,
                              void* d_out, int out_size, void* d_ws, size_t ws_size,
                              hipStream_t stream) {
  const int*   tokens = (const int*)  d_in[0];
  const float* emb    = (const float*)d_in[1];
  const float* pe     = (const float*)d_in[2];
  const float* ln1_g  = (const float*)d_in[3];
  const float* ln1_b  = (const float*)d_in[4];
  const float* Wq     = (const float*)d_in[5];
  const float* bq     = (const float*)d_in[6];
  const float* Wk     = (const float*)d_in[7];
  const float* bk     = (const float*)d_in[8];
  const float* Wv     = (const float*)d_in[9];
  const float* bv     = (const float*)d_in[10];
  const float* Wo     = (const float*)d_in[11];
  const float* bo     = (const float*)d_in[12];
  const float* ln2_g  = (const float*)d_in[13];
  const float* ln2_b  = (const float*)d_in[14];
  const float* W1     = (const float*)d_in[15];
  const float* b1     = (const float*)d_in[16];
  const float* W2     = (const float*)d_in[17];
  const float* b2     = (const float*)d_in[18];
  const float* lnf_g  = (const float*)d_in[19];
  const float* lnf_b  = (const float*)d_in[20];
  const float* fcW    = (const float*)d_in[21];
  const float* fcb    = (const float*)d_in[22];
  float* out = (float*)d_out;

  const size_t MD = (size_t)Mrows * Dm;     // 4.19M
  const size_t MF = (size_t)Mrows * Ffn;    // 16.8M
  const size_t MC = (size_t)Mrows * Cv;     // 33.5M
  float*  x      = (float*)d_ws;            // residual stream (f32)
  float*  logits = x + MD;
  __bf16* hb     = (__bf16*)(logits + MC);  // LN output
  __bf16* qb     = hb + MD;
  __bf16* kb     = qb + MD;
  __bf16* vb     = kb + MD;
  __bf16* yb     = vb + MD;                 // attention out
  __bf16* fb     = yb + MD;                 // FFN hidden
  __bf16* wbf    = fb + MF;                 // weight scratch (max Cv*Dm = 8M)

  const dim3 blk(256);
  const dim3 gD (Dm  / 128, Mrows / 128);   // (8, 32)
  const dim3 gFF(Ffn / 128, Mrows / 128);   // (32, 32)
  const dim3 gC (Cv  / 128, Mrows / 128);   // (64, 32)
  const dim3 gAt(Seq / 128, Bsz * Hn);      // (8, 64)
  const int cDD = (Dm * Dm) / 8 / 256;      // cvt grid for DxD
  const int cDF = (Dm * Ffn) / 8 / 256;
  const int cDC = (Dm * Cv) / 8 / 256;

  embed_kernel<<<Mrows, blk, 0, stream>>>(tokens, emb, pe, x);

  for (int i = 0; i < Ly; ++i) {
    const size_t wD = (size_t)i * Dm * Dm;
    const size_t wF = (size_t)i * (size_t)Ffn * Dm;
    ln_kernel<<<Mrows, blk, 0, stream>>>(x, ln1_g + i * Dm, ln1_b + i * Dm, hb);

    cvt_kernel<<<cDD, blk, 0, stream>>>(Wq + wD, wbf, (Dm * Dm) / 8);
    gemm_nt<<<gD, blk, 0, stream>>>(hb, wbf, bq + i * Dm, nullptr, qb, Mrows, Dm, Dm, GF_BIAS | GF_OBF);
    cvt_kernel<<<cDD, blk, 0, stream>>>(Wk + wD, wbf, (Dm * Dm) / 8);
    gemm_nt<<<gD, blk, 0, stream>>>(hb, wbf, bk + i * Dm, nullptr, kb, Mrows, Dm, Dm, GF_BIAS | GF_OBF);
    cvt_kernel<<<cDD, blk, 0, stream>>>(Wv + wD, wbf, (Dm * Dm) / 8);
    gemm_nt<<<gD, blk, 0, stream>>>(hb, wbf, bv + i * Dm, nullptr, vb, Mrows, Dm, Dm, GF_BIAS | GF_OBF);

    flash_attn<<<gAt, blk, 0, stream>>>(qb, kb, vb, yb);

    cvt_kernel<<<cDD, blk, 0, stream>>>(Wo + wD, wbf, (Dm * Dm) / 8);
    gemm_nt<<<gD, blk, 0, stream>>>(yb, wbf, bo + i * Dm, x, nullptr, Mrows, Dm, Dm, GF_BIAS | GF_RES);

    ln_kernel<<<Mrows, blk, 0, stream>>>(x, ln2_g + i * Dm, ln2_b + i * Dm, hb);

    cvt_kernel<<<cDF, blk, 0, stream>>>(W1 + wF, wbf, (Ffn * Dm) / 8);
    gemm_nt<<<gFF, blk, 0, stream>>>(hb, wbf, b1 + i * Ffn, nullptr, fb, Mrows, Ffn, Dm, GF_BIAS | GF_GELU | GF_OBF);
    cvt_kernel<<<cDF, blk, 0, stream>>>(W2 + wF, wbf, (Dm * Ffn) / 8);
    gemm_nt<<<gD, blk, 0, stream>>>(fb, wbf, b2 + i * Dm, x, nullptr, Mrows, Dm, Ffn, GF_BIAS | GF_RES);
  }

  ln_kernel<<<Mrows, blk, 0, stream>>>(x, lnf_g, lnf_b, hb);
  cvt_kernel<<<cDC, blk, 0, stream>>>(fcW, wbf, (Cv * Dm) / 8);
  gemm_nt<<<gC, blk, 0, stream>>>(hb, wbf, fcb, logits, nullptr, Mrows, Cv, Dm, GF_BIAS);
  zero_out<<<1, 32, 0, stream>>>(out);
  loss_kernel<<<Mrows, blk, 0, stream>>>(logits, tokens, out);
}
